// TransLayer_1219770712643
// MI455X (gfx1250) — matrix operand order repaired and checked by reading, not compiled, not run on hardware
//
#include <hip/hip_runtime.h>
#include <stdint.h>

// ---------------------------------------------------------------------------
// Nystrom attention block for MI455X (gfx1250, wave32, WMMA).
// All heavy matmuls -> v_wmma_f32_16x16x32_bf16 ; pinv Newton-Schulz matmuls
// -> v_wmma_f32_16x16x4_f32 (f32 precision). Intermediates sized to sit in
// the 192MB L2; fragments loaded as 16B vectors per the ISA wave32 layouts.
// GEMM kernels are templated on epilogue-mode and tile layout so the WMMA
// inner loop is branchless (EXEC must be all-ones for WMMA per ISA 7.12).
// ---------------------------------------------------------------------------

typedef __bf16 bf16;
typedef __bf16 v16bf __attribute__((ext_vector_type(16)));
typedef float  v8f   __attribute__((ext_vector_type(8)));
typedef float  v2f   __attribute__((ext_vector_type(2)));
typedef unsigned int vu4 __attribute__((ext_vector_type(4)));

#if defined(__has_builtin)
#if __has_builtin(__builtin_amdgcn_wmma_f32_16x16x4_f32)
#define HAVE_WMMA_F32 1
#endif
#endif

#define B_    4
#define N_    4096
#define D_    512
#define H_    8
#define DH_   64
#define M_    256
#define TOK_  16384     // B_*N_
#define BH_   32        // B_*H_
#define KCONV_ 33
#define PADC_  16
#define SCALE_ 0.125f   // DH^-0.5

// ---------------- fragment loaders (wave32 WMMA layouts, ISA 7.12.2) -------
// A 16x32 bf16: lane m=lane&15 ; halves: k = (i/8)*16 + (lane>>4)*8 + i%8
__device__ __forceinline__ v16bf ldfragA(const bf16* tile, int ld) {
  int lane = threadIdx.x & 31;
  const bf16* p = tile + (size_t)(lane & 15) * ld + ((lane >> 4) << 3);
  union { vu4 u[2]; v16bf v; } un;
  un.u[0] = *(const vu4*)p;
  un.u[1] = *(const vu4*)(p + 16);
  return un.v;
}
// B 32x16 bf16 from B^T [N][K] row-major: lane n=lane&15, k = (i/8)*16 + (lane>>4)*8 + i%8
__device__ __forceinline__ v16bf ldfragB(const bf16* tileT, int ld) {
  int lane = threadIdx.x & 31;
  const bf16* p = tileT + (size_t)(lane & 15) * ld + ((lane >> 4) << 3);
  union { vu4 u[2]; v16bf v; } un;
  un.u[0] = *(const vu4*)p;
  un.u[1] = *(const vu4*)(p + 16);
  return un.v;
}

// ---------------- block reductions (256 threads, uses LDS) -----------------
__device__ __forceinline__ float block_reduce_sum(float v, float* sh) {
  int t = threadIdx.x;
  sh[t] = v; __syncthreads();
  for (int s = 128; s > 0; s >>= 1) { if (t < s) sh[t] += sh[t + s]; __syncthreads(); }
  float r = sh[0]; __syncthreads();
  return r;
}
__device__ __forceinline__ float block_reduce_max(float v, float* sh) {
  int t = threadIdx.x;
  sh[t] = v; __syncthreads();
  for (int s = 128; s > 0; s >>= 1) { if (t < s) sh[t] = fmaxf(sh[t], sh[t + s]); __syncthreads(); }
  float r = sh[0]; __syncthreads();
  return r;
}

// ---------------- LayerNorm + bf16 cast ------------------------------------
__global__ __launch_bounds__(256) void k_ln(const float* __restrict__ x,
                                            const float* __restrict__ gamma,
                                            const float* __restrict__ beta,
                                            bf16* __restrict__ xn) {
  __shared__ float sh[256];
  int row = blockIdx.x;                 // 0..16383
  const float* xr = x + (size_t)row * D_;
  int t = threadIdx.x;
  float a = xr[t], b = xr[t + 256];
  float s = block_reduce_sum(a + b, sh);
  float mu = s * (1.f / D_);
  float da = a - mu, db = b - mu;
  float s2 = block_reduce_sum(da * da + db * db, sh);
  float inv = rsqrtf(s2 * (1.f / D_) + 1e-5f);
  bf16* o = xn + (size_t)row * D_;
  o[t]       = (bf16)(da * inv * gamma[t]       + beta[t]);
  o[t + 256] = (bf16)(db * inv * gamma[t + 256] + beta[t + 256]);
}

// ---------------- transpose/cast helpers -----------------------------------
__global__ void k_castT(const float* __restrict__ src, bf16* __restrict__ dst,
                        int R, int C, long long sB, long long dB) {
  long long idx = (long long)blockIdx.x * 256 + threadIdx.x;
  if (idx >= (long long)R * C) return;
  int r = (int)(idx / C), c = (int)(idx % C);
  dst[(long long)blockIdx.y * dB + (long long)c * R + r] =
      (bf16)src[(long long)blockIdx.y * sB + idx];
}
__global__ void k_transpose_bf(const bf16* __restrict__ src, bf16* __restrict__ dst,
                               int R, int C, long long sB, long long dB) {
  long long idx = (long long)blockIdx.x * 256 + threadIdx.x;
  if (idx >= (long long)R * C) return;
  int r = (int)(idx / C), c = (int)(idx % C);
  dst[(long long)blockIdx.y * dB + (long long)c * R + r] =
      src[(long long)blockIdx.y * sB + idx];
}

// ---------------- generic bf16 WMMA GEMM: C = A[M,K] x (BT[N,K])^T ---------
// TALL==0: block 128x128, 8 waves (4x2), wave tile 32x64 (2x4 WMMA tiles)
// TALL==1: block 256x64,  8 waves (8x1), wave tile 32x64 (2x4 WMMA tiles)
// All launches use exact multiples -> no runtime guards, EXEC stays all-ones.
struct GemmB {
  const bf16* A; const bf16* BT;
  long long aB, bB;
  int lda, ldbt, M, N, K;
  float alpha;
  float* Cf; bf16* Cb; long long cB; int ldc;
  bf16 *q, *k, *v;                                  // mode 2
  const float* x; const float* bias; float* outf;   // mode 3
};

template <int MODE, int TALL>
__global__ __launch_bounds__(256) void k_gemm_bf16(GemmB g) {
  const int lane = threadIdx.x & 31;
  const int wid  = threadIdx.x >> 5;
  const int wm = TALL ? wid : (wid >> 1);
  const int wn = TALL ? 0   : (wid & 1);
  const int rowBase = blockIdx.x * (TALL ? 256 : 128) + wm * 32;
  const int colBase = blockIdx.y * 128 + wn * 64;
  const long long batch = blockIdx.z;
  const bf16* A  = g.A  + batch * g.aB;
  const bf16* BT = g.BT + batch * g.bB;
  const int lda = g.lda, ldbt = g.ldbt;

  v8f acc[2][4] = {};
  // K is a multiple of 64 in all launches: unroll the 32-wide WMMA step x2 so
  // the compiler can overlap fragment loads with the previous WMMAs.
  for (int k0 = 0; k0 < g.K; k0 += 64) {
    v16bf a0[2], b0[4], a1[2], b1[4];
#pragma unroll
    for (int i = 0; i < 2; ++i) {
      const bf16* ap = A + (size_t)(rowBase + i * 16) * lda + k0;
      a0[i] = ldfragA(ap, lda);
      a1[i] = ldfragA(ap + 32, lda);
    }
#pragma unroll
    for (int j = 0; j < 4; ++j) {
      const bf16* bp = BT + (size_t)(colBase + j * 16) * ldbt + k0;
      b0[j] = ldfragB(bp, ldbt);
      b1[j] = ldfragB(bp + 32, ldbt);
    }
#pragma unroll
    for (int i = 0; i < 2; ++i)
#pragma unroll
      for (int j = 0; j < 4; ++j)
        acc[i][j] = __builtin_amdgcn_wmma_f32_16x16x32_bf16(
            false, a0[i], false, b0[j], (short)0, acc[i][j], false, false);
#pragma unroll
    for (int i = 0; i < 2; ++i)
#pragma unroll
      for (int j = 0; j < 4; ++j)
        acc[i][j] = __builtin_amdgcn_wmma_f32_16x16x32_bf16(
            false, a1[i], false, b1[j], (short)0, acc[i][j], false, false);
  }

#pragma unroll
  for (int i = 0; i < 2; ++i) {
#pragma unroll
    for (int j = 0; j < 4; ++j) {
      int c  = colBase + j * 16 + (lane & 15);
      int r0 = rowBase + i * 16 + ((lane >> 4) << 3);
#pragma unroll
      for (int r = 0; r < 8; ++r) {
        int row = r0 + r;
        float val = acc[i][j][r] * g.alpha;
        if (MODE == 0) {
          g.Cf[batch * g.cB + (long long)row * g.ldc + c] = val;
        } else if (MODE == 1) {
          g.Cb[batch * g.cB + (long long)row * g.ldc + c] = (bf16)val;
        } else if (MODE == 2) {
          int which = c / D_;
          int hd = c - which * D_;
          int h = hd >> 6, d = hd & 63;
          int bb = row >> 12, n = row & (N_ - 1);
          long long dst = ((((long long)(bb * H_ + h)) * N_ + n) << 6) + d;
          bf16* dp = (which == 0) ? g.q : (which == 1) ? g.k : g.v;
          float sc = (which == 0) ? SCALE_ : 1.f;
          dp[dst] = (bf16)(val * sc);
        } else {  // MODE 3: out = x + C + bias
          long long idx = (long long)row * D_ + c;
          g.outf[idx] = g.x[idx] + val + g.bias[c];
        }
      }
    }
  }
}

// ---------------- f32 WMMA GEMM (Newton-Schulz): C = A[M,K] x B[K,N] -------
struct GemmF {
  const float* A; const float* B;
  long long aB, bB;
  int lda, ldb, M, N, K;
  float alpha;
  float* C; long long cB; int ldc;
};

__global__ __launch_bounds__(256) void k_gemm_f32(GemmF g) {
  const long long batch = blockIdx.z;
  const float* A = g.A + batch * g.aB;
  const float* B = g.B + batch * g.bB;
  float* C = g.C + batch * g.cB;
#if HAVE_WMMA_F32
  const int lane = threadIdx.x & 31;
  const int wid  = threadIdx.x >> 5;
  const int wm = wid >> 1, wn = wid & 1;
  const int rowBase = blockIdx.x * 128 + wm * 32;
  const int colBase = blockIdx.y * 64  + wn * 32;
  v8f acc[2][2] = {};
  for (int k0 = 0; k0 < g.K; k0 += 8) {   // 2 unrolled WMMA k-steps of 4
#pragma unroll
    for (int kk = 0; kk < 2; ++kk) {
      int klo = k0 + kk * 4 + ((lane >> 4) << 1);
      v2f af[2], bfr[2];
#pragma unroll
      for (int i = 0; i < 2; ++i) {
        const float* p = A + (size_t)(rowBase + i * 16 + (lane & 15)) * g.lda + klo;
        af[i] = *(const v2f*)p;
      }
#pragma unroll
      for (int j = 0; j < 2; ++j) {
        int col = colBase + j * 16 + (lane & 15);
        v2f b;
        b.x = B[(size_t)klo * g.ldb + col];
        b.y = B[(size_t)(klo + 1) * g.ldb + col];
        bfr[j] = b;
      }
#pragma unroll
      for (int i = 0; i < 2; ++i)
#pragma unroll
        for (int j = 0; j < 2; ++j)
          acc[i][j] = __builtin_amdgcn_wmma_f32_16x16x4_f32(
              false, af[i], false, bfr[j], (short)0, acc[i][j], false, false);
    }
  }
#pragma unroll
  for (int i = 0; i < 2; ++i)
#pragma unroll
    for (int j = 0; j < 2; ++j) {
      int c  = colBase + j * 16 + (lane & 15);
      int r0 = rowBase + i * 16 + ((lane >> 4) << 3);
#pragma unroll
      for (int r = 0; r < 8; ++r)
        C[(size_t)(r0 + r) * g.ldc + c] = acc[i][j][r] * g.alpha;
    }
#else
  // scalar fallback (kept correct if the f32 WMMA builtin is absent)
  int bm0 = blockIdx.x * 128, bn0 = blockIdx.y * 64;
  for (int e = threadIdx.x; e < 128 * 64; e += 256) {
    int row = bm0 + (e >> 6), col = bn0 + (e & 63);
    float s = 0.f;
    for (int k = 0; k < g.K; ++k) s += A[(size_t)row * g.lda + k] * B[(size_t)k * g.ldb + col];
    C[(size_t)row * g.ldc + col] = s * g.alpha;
  }
#endif
}

// ---------------- row softmax (in-place, f32 math) -------------------------
template <typename T>
__global__ __launch_bounds__(256) void k_softmax_rows(T* __restrict__ buf, int cols) {
  __shared__ float sh[256];
  T* p = buf + (size_t)blockIdx.x * cols;
  int t = threadIdx.x;
  float m = -3.4e38f;
  for (int c = t; c < cols; c += 256) m = fmaxf(m, (float)p[c]);
  m = block_reduce_max(m, sh);
  float s = 0.f;
  for (int c = t; c < cols; c += 256) {
    float e = __expf((float)p[c] - m);
    p[c] = (T)e;
    s += e;
  }
  s = block_reduce_sum(s, sh);
  float inv = 1.f / s;
  for (int c = t; c < cols; c += 256) p[c] = (T)((float)p[c] * inv);
}

// ---------------- landmark means (l = 16) ----------------------------------
__global__ __launch_bounds__(64) void k_landmark(const bf16* __restrict__ q,
                                                 const bf16* __restrict__ k,
                                                 bf16* __restrict__ ql,
                                                 bf16* __restrict__ kl) {
  int bm = blockIdx.x;                  // bh*256 + m
  int bh = bm >> 8, m = bm & 255;
  int d = threadIdx.x;
  size_t base = ((size_t)bh * N_ + (size_t)m * 16) * DH_ + d;
  float sq = 0.f, sk = 0.f;
  for (int l = 0; l < 16; ++l) {
    sq += (float)q[base + (size_t)l * DH_];
    sk += (float)k[base + (size_t)l * DH_];
  }
  size_t o = ((size_t)bh * M_ + m) * DH_ + d;
  ql[o] = (bf16)(sq * (1.f / 16.f));
  kl[o] = (bf16)(sk * (1.f / 16.f));
}

// ---------------- pinv init: global max of |row sums| / |col sums| ---------
__global__ void k_zero2(unsigned* s) { if (threadIdx.x < 2) s[threadIdx.x] = 0u; }

__global__ __launch_bounds__(256) void k_absmax(const float* __restrict__ a2, unsigned* mx) {
  __shared__ float sh[256];
  int bh = blockIdx.x, j = threadIdx.x;
  const float* A = a2 + (size_t)bh * (M_ * M_);
  float cs = 0.f, rs = 0.f;
  for (int i = 0; i < M_; ++i) {
    cs += fabsf(A[(size_t)i * M_ + j]);   // column sums
    rs += fabsf(A[(size_t)j * M_ + i]);   // row sums
  }
  float cm = block_reduce_max(cs, sh);
  float rm = block_reduce_max(rs, sh);
  if (j == 0) {
    atomicMax(&mx[0], __float_as_uint(cm));
    atomicMax(&mx[1], __float_as_uint(rm));
  }
}

__global__ void k_zinit(const float* __restrict__ a2, float* __restrict__ z,
                        const unsigned* __restrict__ mx) {
  float rcp = 1.f / (__uint_as_float(mx[0]) * __uint_as_float(mx[1]));
  int idx = blockIdx.x * 256 + threadIdx.x;   // 0..65535
  int i = idx >> 8, j = idx & 255;
  size_t base = (size_t)blockIdx.y * (M_ * M_);
  z[base + idx] = a2[base + (size_t)j * M_ + i] * rcp;   // z = a2^T * rcp
}

__global__ void k_diag_sub(const float* __restrict__ P, float* __restrict__ Y, float c) {
  int idx = blockIdx.x * 256 + threadIdx.x;
  int i = idx >> 8, j = idx & 255;
  size_t base = (size_t)blockIdx.y * (M_ * M_);
  float v = -P[base + idx];
  if (i == j) v += c;
  Y[base + idx] = v;        // Y = c*I - P
}

// ---------------- depthwise residual conv + head merge ---------------------
__global__ __launch_bounds__(256) void k_conv_merge(const float* __restrict__ oh,
                                                    const bf16* __restrict__ v,
                                                    const float* __restrict__ rw,
                                                    bf16* __restrict__ obf) {
  long long idx = (long long)blockIdx.x * 256 + threadIdx.x;   // [b,h,n,d] flat
  int d = (int)(idx & 63);
  int n = (int)((idx >> 6) & (N_ - 1));
  int h = (int)((idx >> 18) & (H_ - 1));
  int b = (int)(idx >> 21);
  float acc = oh[idx];
  size_t vbase = ((size_t)(b * H_ + h) * N_) * DH_ + d;
  for (int t = 0; t < KCONV_; ++t) {
    int nn = n + t - PADC_;
    if (nn >= 0 && nn < N_) acc += rw[h * KCONV_ + t] * (float)v[vbase + (size_t)nn * DH_];
  }
  obf[((size_t)(b * N_ + n)) * D_ + h * DH_ + d] = (bf16)acc;
}

// ---------------------------------------------------------------------------
extern "C" void kernel_launch(void* const* d_in, const int* in_sizes, int n_in,
                              void* d_out, int out_size, void* d_ws, size_t ws_size,
                              hipStream_t stream) {
  const float* x     = (const float*)d_in[0];
  const float* gamma = (const float*)d_in[1];
  const float* beta  = (const float*)d_in[2];
  const float* wqkv  = (const float*)d_in[3];
  const float* wout  = (const float*)d_in[4];
  const float* bout  = (const float*)d_in[5];
  const float* resw  = (const float*)d_in[6];
  float* out = (float*)d_out;

  // ---- workspace carve (~261 MB total) ----
  char* w = (char*)d_ws;
  size_t off = 0;
  auto alloc = [&](size_t bytes) -> void* {
    void* p = w + off;
    off = (off + bytes + 255) & ~(size_t)255;
    return p;
  };
  bf16* xn_bf  = (bf16*)alloc((size_t)TOK_ * D_ * 2);
  bf16* wqkvT  = (bf16*)alloc((size_t)3 * D_ * D_ * 2);
  bf16* woutT  = (bf16*)alloc((size_t)D_ * D_ * 2);
  bf16* q_bf   = (bf16*)alloc((size_t)BH_ * N_ * DH_ * 2);
  bf16* k_bf   = (bf16*)alloc((size_t)BH_ * N_ * DH_ * 2);
  bf16* v_bf   = (bf16*)alloc((size_t)BH_ * N_ * DH_ * 2);
  bf16* vT_bf  = (bf16*)alloc((size_t)BH_ * N_ * DH_ * 2);
  bf16* ql_bf  = (bf16*)alloc((size_t)BH_ * M_ * DH_ * 2);
  bf16* kl_bf  = (bf16*)alloc((size_t)BH_ * M_ * DH_ * 2);
  float* a2    = (float*)alloc((size_t)BH_ * M_ * M_ * 4);
  float* z0    = (float*)alloc((size_t)BH_ * M_ * M_ * 4);
  float* z1    = (float*)alloc((size_t)BH_ * M_ * M_ * 4);
  float* xz    = (float*)alloc((size_t)BH_ * M_ * M_ * 4);
  float* ybuf  = (float*)alloc((size_t)BH_ * M_ * M_ * 4);
  float* pbuf  = (float*)alloc((size_t)BH_ * M_ * M_ * 4);
  float* tbuf  = (float*)alloc((size_t)BH_ * M_ * DH_ * 4);
  float* ubuf  = (float*)alloc((size_t)BH_ * M_ * DH_ * 4);
  bf16* uT_bf  = (bf16*)alloc((size_t)BH_ * M_ * DH_ * 2);
  bf16* s_buf  = (bf16*)alloc((size_t)BH_ * N_ * M_ * 2);   // reused: a3 then a1
  float* ohbuf = (float*)alloc((size_t)BH_ * N_ * DH_ * 4);
  bf16* o_bf   = (bf16*)alloc((size_t)TOK_ * D_ * 2);
  unsigned* mx = (unsigned*)alloc(8);
  (void)ws_size; (void)n_in; (void)in_sizes; (void)out_size;

  // 1) LayerNorm + cast
  k_ln<<<TOK_, 256, 0, stream>>>(x, gamma, beta, xn_bf);

  // 2) weight transpose+cast: B^T layouts for WMMA B fragments
  k_castT<<<dim3((3 * D_ * D_ + 255) / 256, 1), 256, 0, stream>>>(wqkv, wqkvT, D_, 3 * D_, 0, 0);
  k_castT<<<dim3((D_ * D_ + 255) / 256, 1), 256, 0, stream>>>(wout, woutT, D_, D_, 0, 0);

  // 3) QKV GEMM (mode 2 scatters into q(*scale)/k/v per-head layout)
  {
    GemmB g = {};
    g.A = xn_bf; g.aB = 0; g.lda = D_;
    g.BT = wqkvT; g.bB = 0; g.ldbt = D_;
    g.M = TOK_; g.N = 3 * D_; g.K = D_;
    g.alpha = 1.f;
    g.q = q_bf; g.k = k_bf; g.v = v_bf;
    k_gemm_bf16<2, 0><<<dim3(TOK_ / 128, (3 * D_) / 128, 1), 256, 0, stream>>>(g);
  }

  // 4) landmark means
  k_landmark<<<BH_ * M_, 64, 0, stream>>>(q_bf, k_bf, ql_bf, kl_bf);

  // 5) sim2 = ql @ kl^T  -> f32, then softmax -> a2
  {
    GemmB g = {};
    g.A = ql_bf; g.aB = (long long)M_ * DH_; g.lda = DH_;
    g.BT = kl_bf; g.bB = (long long)M_ * DH_; g.ldbt = DH_;
    g.M = M_; g.N = M_; g.K = DH_;
    g.alpha = 1.f;
    g.Cf = a2; g.cB = (long long)M_ * M_; g.ldc = M_;
    k_gemm_bf16<0, 0><<<dim3(2, 2, BH_), 256, 0, stream>>>(g);
  }
  k_softmax_rows<float><<<BH_ * M_, 256, 0, stream>>>(a2, M_);

  // 6) pinv init: z = a2^T / (max(rowsums)*max(colsums))  (global maxes)
  k_zero2<<<1, 32, 0, stream>>>(mx);
  k_absmax<<<BH_, 256, 0, stream>>>(a2, mx);
  k_zinit<<<dim3((M_ * M_) / 256, BH_), 256, 0, stream>>>(a2, z0, mx);

  // 7) Newton-Schulz: z = 0.25*z@(13I - xz@(15I - xz@(7I - xz))), xz = a2@z
  float* zcur = z0;
  float* znxt = z1;
  auto gemmF = [&](const float* A, const float* Bm, float* C, int Mr, int Nr, int Kr,
                   int lda, int ldb, int ldc, long long aB, long long bB, long long cB,
                   float alpha) {
    GemmF g = { A, Bm, aB, bB, lda, ldb, Mr, Nr, Kr, alpha, C, cB, ldc };
    k_gemm_f32<<<dim3(Mr / 128, Nr / 64, BH_), 256, 0, stream>>>(g);
  };
  const long long MM = (long long)M_ * M_;
  for (int it = 0; it < 6; ++it) {
    gemmF(a2, zcur, xz, M_, M_, M_, M_, M_, M_, MM, MM, MM, 1.f);
    k_diag_sub<<<dim3((M_ * M_) / 256, BH_), 256, 0, stream>>>(xz, ybuf, 7.f);
    gemmF(xz, ybuf, pbuf, M_, M_, M_, M_, M_, M_, MM, MM, MM, 1.f);
    k_diag_sub<<<dim3((M_ * M_) / 256, BH_), 256, 0, stream>>>(pbuf, ybuf, 15.f);
    gemmF(xz, ybuf, pbuf, M_, M_, M_, M_, M_, M_, MM, MM, MM, 1.f);
    k_diag_sub<<<dim3((M_ * M_) / 256, BH_), 256, 0, stream>>>(pbuf, ybuf, 13.f);
    gemmF(zcur, ybuf, znxt, M_, M_, M_, M_, M_, M_, MM, MM, MM, 0.25f);
    float* tmp = zcur; zcur = znxt; znxt = tmp;
  }

  // 8) sim3 = ql @ k^T -> bf16 s_buf [bh,256,4096]; softmax over n
  {
    GemmB g = {};
    g.A = ql_bf; g.aB = (long long)M_ * DH_; g.lda = DH_;
    g.BT = k_bf; g.bB = (long long)N_ * DH_; g.ldbt = DH_;
    g.M = M_; g.N = N_; g.K = DH_;
    g.alpha = 1.f;
    g.Cb = s_buf; g.cB = (long long)M_ * N_; g.ldc = N_;
    k_gemm_bf16<1, 0><<<dim3(2, N_ / 128, BH_), 256, 0, stream>>>(g);
  }
  k_softmax_rows<bf16><<<BH_ * M_, 256, 0, stream>>>(s_buf, N_);

  // 9) t = a3 @ v   (v^T staged once; TALL layout for N=64)
  k_transpose_bf<<<dim3((N_ * DH_) / 256, BH_), 256, 0, stream>>>(
      v_bf, vT_bf, N_, DH_, (long long)N_ * DH_, (long long)N_ * DH_);
  {
    GemmB g = {};
    g.A = s_buf; g.aB = (long long)M_ * N_; g.lda = N_;
    g.BT = vT_bf; g.bB = (long long)N_ * DH_; g.ldbt = N_;
    g.M = M_; g.N = DH_; g.K = N_;
    g.alpha = 1.f;
    g.Cf = tbuf; g.cB = (long long)M_ * DH_; g.ldc = DH_;
    k_gemm_bf16<0, 1><<<dim3(M_ / 256, 1, BH_), 256, 0, stream>>>(g);
  }

  // 10) u = a2inv @ t (f32 WMMA), then u^T -> bf16
  gemmF(zcur, tbuf, ubuf, M_, DH_, M_, M_, DH_, DH_, MM, (long long)M_ * DH_,
        (long long)M_ * DH_, 1.f);
  k_castT<<<dim3((M_ * DH_) / 256, BH_), 256, 0, stream>>>(
      ubuf, uT_bf, M_, DH_, (long long)M_ * DH_, (long long)M_ * DH_);

  // 11) sim1 = q @ kl^T -> s_buf [bh,4096,256]; softmax over M
  {
    GemmB g = {};
    g.A = q_bf; g.aB = (long long)N_ * DH_; g.lda = DH_;
    g.BT = kl_bf; g.bB = (long long)M_ * DH_; g.ldbt = DH_;
    g.M = N_; g.N = M_; g.K = DH_;
    g.alpha = 1.f;
    g.Cb = s_buf; g.cB = (long long)N_ * M_; g.ldc = M_;
    k_gemm_bf16<1, 0><<<dim3(N_ / 128, 2, BH_), 256, 0, stream>>>(g);
  }
  k_softmax_rows<bf16><<<BH_ * N_, 256, 0, stream>>>(s_buf, M_);

  // 12) oh = a1 @ u -> f32 [bh,4096,64]  (TALL layout for N=64)
  {
    GemmB g = {};
    g.A = s_buf; g.aB = (long long)N_ * M_; g.lda = M_;
    g.BT = uT_bf; g.bB = (long long)M_ * DH_; g.ldbt = M_;
    g.M = N_; g.N = DH_; g.K = M_;
    g.alpha = 1.f;
    g.Cf = ohbuf; g.cB = (long long)N_ * DH_; g.ldc = DH_;
    k_gemm_bf16<0, 1><<<dim3(N_ / 256, 1, BH_), 256, 0, stream>>>(g);
  }

  // 13) depthwise residual conv + head merge -> o_bf [16384,512]
  k_conv_merge<<<(BH_ * N_ * DH_) / 256, 256, 0, stream>>>(ohbuf, v_bf, resw, o_bf);

  // 14) final: out = x + o @ w_out + b_out  (fused epilogue)
  {
    GemmB g = {};
    g.A = o_bf; g.aB = 0; g.lda = D_;
    g.BT = woutT; g.bB = 0; g.ldbt = D_;
    g.M = TOK_; g.N = D_; g.K = D_;
    g.alpha = 1.f;
    g.x = x; g.bias = bout; g.outf = out;
    k_gemm_bf16<3, 0><<<dim3(TOK_ / 128, D_ / 128, 1), 256, 0, stream>>>(g);
  }
}
